// MiddleLayerEncoder_27350351741101
// MI455X (gfx1250) — matrix operand order repaired and checked by reading, not compiled, not run on hardware
//
#include <hip/hip_runtime.h>
#include <hip/hip_bf16.h>

typedef __attribute__((ext_vector_type(16))) _Float16 v16h;
typedef __attribute__((ext_vector_type(8)))  _Float16 v8h;
typedef __attribute__((ext_vector_type(4)))  _Float16 v4h;
typedef __attribute__((ext_vector_type(2)))  _Float16 v2h;
typedef __attribute__((ext_vector_type(8)))  float    v8f;

#define PTS      16
#define WAVES    8
#define BLOCK    (WAVES * 32)

// ---------------- LDS layout (units: halfs unless noted) ----------------
// Weights packed per 32x16 WMMA tile: 512 halfs/tile, read as v16h per lane.
#define W_ENC1   0            // 1 kt x 2 nt  (K=32 pad of 3,  N=32)
#define W_ENC2   1024         // 1 kt x 4 nt  (K=32,           N=64)
#define W_FC1    3072         // 5 kt x 8 nt  (K=160 pad 131,  N=128)
#define W_FC2    23552        // 4 kt x 8 nt  (K=128,          N=128)
#define W_G1     39936        // 4 kt x 8 nt  (K=128,          N=128)
#define W_G2     56320        // 4 kt x 16 nt (K=128,          N=256)
#define W_TOTAL  89088

#define ACT_PITCH 168                           // 336 B rows: 16B aligned, conflict-free
#define ACT_SIZE  (16 * ACT_PITCH)              // 2688 halfs
#define ACT_OFF   W_TOTAL                       // per-wave [16][168] activation scratch
#define IN_OFF    (ACT_OFF + WAVES * ACT_SIZE)  // per-wave [16][168] fc1-input staging
#define NB_OFF    (IN_OFF + WAVES * ACT_SIZE)   // per-wave nb_enc[64]
#define GA_OFF    (NB_OFF + WAVES * 64)         // group A-tile for g-MLP: 4 kt x 512
#define H_TOTAL   (GA_OFF + 2048)
#define SMEM_BYTES (H_TOTAL * 2)

// fc1-input column order (chosen for vectorized LDS access):
//   cols 0-63   : dup (nb_enc broadcast)   <- fw1 rows 67..130
//   cols 64-127 : features                 <- fw1 rows 3..66
//   cols 128-130: relative_points          <- fw1 rows 0..2
//   cols 131-159: zero pad

// ---- A-operand (16-bit, 16x32) lane/half mapping per CDNA5 ISA ----
__device__ __forceinline__ int a_slot(int kk, int mm) {
    int lane, h;
    if (kk < 8)       { lane = mm;      h = kk;      }
    else if (kk < 16) { lane = mm + 16; h = kk - 8;  }
    else if (kk < 24) { lane = mm;      h = kk - 8;  }
    else              { lane = mm + 16; h = kk - 16; }
    return lane * 16 + h;
}
// ---- B-operand (16-bit, 32x16): lanes 0-15 K=0-7,16-23; lanes 16-31 K=8-15,24-31
__device__ __forceinline__ int b_slot(int kk, int nn) {
    return (nn + (((kk >> 3) & 1) << 4)) * 16 + (kk & 7) + ((kk >> 4) << 3);
}

__device__ __forceinline__ v8f wmma_(v16h a, v16h b, v8f c) {
    return __builtin_amdgcn_wmma_f32_16x16x32_f16(false, a, false, b, (short)0, c,
                                                  false, false);
}

// Lane's A operand for one 16x32 K-tile: two contiguous 8-half (16B) runs.
__device__ __forceinline__ v16h read_a_row(const _Float16* row, int hi) {
    v8h lo = *(const v8h*)(row + (hi ? 8 : 0));        // kk 0-7  / 8-15
    v8h up = *(const v8h*)(row + 16 + (hi ? 8 : 0));   // kk 16-23 / 24-31
    v16h a;
    #pragma unroll
    for (int i = 0; i < 8; ++i) { a[i] = lo[i]; a[i + 8] = up[i]; }
    return a;
}

__device__ __forceinline__ void pack_weight(_Float16* lh, int dst, const float* src,
                                            int K, int Nf, int NT) {
    for (int idx = threadIdx.x; idx < K * Nf; idx += BLOCK) {
        int k = idx / Nf, n = idx % Nf;
        lh[dst + ((k >> 5) * NT + (n >> 4)) * 512 + b_slot(k & 31, n & 15)] =
            (_Float16)src[idx];
    }
}
// fc1 pack with the input-column permutation above
__device__ __forceinline__ void pack_fc1(_Float16* lh, const float* src) {
    for (int idx = threadIdx.x; idx < 131 * 128; idx += BLOCK) {
        int r = idx / 128, n = idx % 128;
        int col = (r < 3) ? (128 + r) : (r < 67 ? (61 + r) : (r - 67));
        lh[W_FC1 + ((col >> 5) * 8 + (n >> 4)) * 512 + b_slot(col & 31, n & 15)] =
            (_Float16)src[idx];
    }
}

__global__ __launch_bounds__(BLOCK, 1)
void mle_fused(const float* __restrict__ rel, const float* __restrict__ feat,
               const float* __restrict__ ew1, const float* __restrict__ eb1,
               const float* __restrict__ ew2, const float* __restrict__ eb2,
               const float* __restrict__ fw1, const float* __restrict__ fb1,
               const float* __restrict__ fw2, const float* __restrict__ fb2,
               const float* __restrict__ gw1, const float* __restrict__ gb1,
               const float* __restrict__ gw2, const float* __restrict__ gb2,
               float* __restrict__ out, int NC, int ngroups)
{
    extern __shared__ char smem[];
    _Float16* lh = (_Float16*)smem;

    const int tid  = threadIdx.x;
    const int lane = tid & 31;
    const int wid  = tid >> 5;
    const int lm   = lane & 15;
    const int hi   = lane >> 4;

    // ---- one-time: pack all weights (f32 -> f16, WMMA-B swizzled) into LDS ----
    {
        v8h zz = {};
        for (int i = tid; i < W_TOTAL / 8; i += BLOCK) *(v8h*)(lh + i * 8) = zz;
    }
    __syncthreads();
    pack_weight(lh, W_ENC1, ew1,   3,  32,  2);
    pack_weight(lh, W_ENC2, ew2,  32,  64,  4);
    pack_fc1(lh, fw1);
    pack_weight(lh, W_FC2,  fw2, 128, 128,  8);
    pack_weight(lh, W_G1,   gw1, 128, 128,  8);
    pack_weight(lh, W_G2,   gw2, 128, 256, 16);

    // ---- hoist all biases into per-lane registers (kernel-invariant) ----
    float bE1[2], bE2[4], bF1[8], bF2[8], bG1[8], bG2x0, bG2x1;
    #pragma unroll
    for (int nt = 0; nt < 2; ++nt) bE1[nt] = eb1[nt * 16 + lm];
    #pragma unroll
    for (int nt = 0; nt < 4; ++nt) bE2[nt] = eb2[nt * 16 + lm];
    #pragma unroll
    for (int nt = 0; nt < 8; ++nt) bF1[nt] = fb1[nt * 16 + lm];
    #pragma unroll
    for (int nt = 0; nt < 8; ++nt) bF2[nt] = fb2[nt * 16 + lm];
    #pragma unroll
    for (int nt = 0; nt < 8; ++nt) bG1[nt] = gb1[nt * 16 + lm];
    bG2x0 = gb2[(wid * 2 + 0) * 16 + lm];
    bG2x1 = gb2[(wid * 2 + 1) * 16 + lm];

    _Float16* act  = lh + ACT_OFF + wid * ACT_SIZE;   // this wave's [16][168]
    _Float16* act0 = lh + ACT_OFF;                    // wave 0's region (g1 output)
    _Float16* in   = lh + IN_OFF + wid * ACT_SIZE;    // this wave's fc1-input staging
    _Float16* nb   = lh + NB_OFF + wid * 64;          // this wave's nb_enc[64]

    // zero pad columns 128..167 of the staging buffer once (rel cols 128-130 are
    // rewritten every cluster; 131-159 must stay zero for enc1/fc1 kt=4)
    {
        v8h zz = {};
        for (int i = lane; i < 16 * 5; i += 32) {      // 5 x v8h per row
            int r = i / 5, ch = i % 5;
            *(v8h*)(in + r * ACT_PITCH + 128 + ch * 8) = zz;
        }
    }
    __syncthreads();

    for (int g = blockIdx.x; g < ngroups; g += gridDim.x) {
        {
            v8h zz = {};
            for (int i = tid; i < 256; i += BLOCK)     // gA: 2048 halfs
                *(v8h*)(lh + GA_OFF + i * 8) = zz;
        }
        __syncthreads();

        // -------- per-point phase: each wave processes 2 clusters --------
        for (int cl = 0; cl < 2; ++cl) {
            const int mc = wid * 2 + cl;        // cluster slot in [16x128] group tile
            const int c  = g * 16 + mc;
            if (c < NC) {
                // ---- stage features: coalesced b128 loads -> f16 LDS rows ----
                const float4* f4 = (const float4*)(feat + (size_t)c * PTS * 64);
                __builtin_prefetch(f4 + 256, 0, 0);        // next cluster's features
                #pragma unroll
                for (int it = 0; it < 8; ++it) {
                    int idx4 = lane + it * 32;             // 0..255 (16 float4 / row)
                    float4 v = f4[idx4];
                    int m = idx4 >> 4, j = (idx4 & 15) << 2;
                    v4h p;
                    p[0] = (_Float16)v.x; p[1] = (_Float16)v.y;
                    p[2] = (_Float16)v.z; p[3] = (_Float16)v.w;
                    *(v4h*)(in + m * ACT_PITCH + 64 + j) = p;
                }
                // ---- stage rel: branchless; lane lm owns its point's 3 floats ----
                {
                    const float* rp = rel + ((size_t)c * PTS + lm) * 3;
                    float x = rp[0], y = rp[1], zv = rp[2];
                    if (hi == 0) {
                        _Float16* dst = in + lm * ACT_PITCH + 128;
                        v2h p2; p2[0] = (_Float16)x; p2[1] = (_Float16)y;
                        *(v2h*)dst = p2;            // b32 store (cols 128-129)
                        dst[2] = (_Float16)zv;      // b16 store (col 130)
                    }
                }
                v8f z = {};
                // ---- enc1: [16x3(pad32)] x [3x32]; A = staging cols 128-159 ----
                {
                    v16h a1 = read_a_row(in + lm * ACT_PITCH + 128, hi);
                    #pragma unroll
                    for (int nt = 0; nt < 2; ++nt) {
                        v16h b = *(const v16h*)(lh + W_ENC1 + nt * 512 + lane * 16);
                        v8f d  = wmma_(a1, b, z);
                        #pragma unroll
                        for (int r = 0; r < 8; ++r)
                            act[(r + hi * 8) * ACT_PITCH + nt * 16 + lm] =
                                (_Float16)fmaxf(d[r] + bE1[nt], 0.f);
                    }
                }
                // ---- enc2: [16x32] x [32x64], per-cluster column max -> nb ----
                {
                    v16h a2 = read_a_row(act + lm * ACT_PITCH, hi);
                    #pragma unroll
                    for (int nt = 0; nt < 4; ++nt) {
                        v16h b = *(const v16h*)(lh + W_ENC2 + nt * 512 + lane * 16);
                        v8f d  = wmma_(a2, b, z);
                        float mv = 0.f;                    // relu output >= 0
                        #pragma unroll
                        for (int r = 0; r < 8; ++r)
                            mv = fmaxf(mv, fmaxf(d[r] + bE2[nt], 0.f));
                        mv = fmaxf(mv, __shfl_xor(mv, 16, 32));
                        if (lane < 16) nb[nt * 16 + lane] = (_Float16)mv;
                    }
                }
                // ---- replicate nb into dup cols 0-63 of all 16 rows (b128) ----
                {
                    v8h chunk = *(const v8h*)(nb + (lane & 7) * 8);
                    int rowbase = (lane >> 3) * 4;
                    #pragma unroll
                    for (int rr = 0; rr < 4; ++rr)
                        *(v8h*)(in + (rowbase + rr) * ACT_PITCH + (lane & 7) * 8) =
                            chunk;
                }
                // ---- fc1: [16x131(pad160)] x [131x128]; pure vector LDS A ----
                v8f e1[8];
                #pragma unroll
                for (int nt = 0; nt < 8; ++nt) e1[nt] = z;
                #pragma unroll
                for (int kt = 0; kt < 5; ++kt) {
                    v16h af = read_a_row(in + lm * ACT_PITCH + kt * 32, hi);
                    #pragma unroll
                    for (int nt = 0; nt < 8; ++nt) {
                        v16h b = *(const v16h*)(lh + W_FC1 + (kt * 8 + nt) * 512 + lane * 16);
                        e1[nt] = wmma_(af, b, e1[nt]);
                    }
                }
                #pragma unroll
                for (int nt = 0; nt < 8; ++nt) {
                    #pragma unroll
                    for (int r = 0; r < 8; ++r)
                        act[(r + hi * 8) * ACT_PITCH + nt * 16 + lm] =
                            (_Float16)fmaxf(e1[nt][r] + bF1[nt], 0.f);
                }
                // ---- fc2: [16x128] x [128x128], then per-cluster max -> gA ----
                v8f e2[8];
                #pragma unroll
                for (int nt = 0; nt < 8; ++nt) e2[nt] = z;
                #pragma unroll
                for (int kt = 0; kt < 4; ++kt) {
                    v16h af = read_a_row(act + lm * ACT_PITCH + kt * 32, hi);
                    #pragma unroll
                    for (int nt = 0; nt < 8; ++nt) {
                        v16h b = *(const v16h*)(lh + W_FC2 + (kt * 8 + nt) * 512 + lane * 16);
                        e2[nt] = wmma_(af, b, e2[nt]);
                    }
                }
                #pragma unroll
                for (int nt = 0; nt < 8; ++nt) {
                    float mv = 0.f;
                    #pragma unroll
                    for (int r = 0; r < 8; ++r)
                        mv = fmaxf(mv, fmaxf(e2[nt][r] + bF2[nt], 0.f));
                    mv = fmaxf(mv, __shfl_xor(mv, 16, 32));
                    if (lane < 16) {
                        int kcol = nt * 16 + lane;       // g-row column (K of g1)
                        lh[GA_OFF + (kcol >> 5) * 512 + a_slot(kcol & 31, mc)] =
                            (_Float16)mv;
                    }
                }
            }
        }
        __syncthreads();

        // -------- global MLP: g1 on wave 0 --------
        if (wid == 0) {
            v8f z = {};
            v8f ga[8];
            #pragma unroll
            for (int nt = 0; nt < 8; ++nt) ga[nt] = z;
            #pragma unroll
            for (int kt = 0; kt < 4; ++kt) {
                v16h a = *(const v16h*)(lh + GA_OFF + kt * 512 + lane * 16);
                #pragma unroll
                for (int nt = 0; nt < 8; ++nt) {
                    v16h b = *(const v16h*)(lh + W_G1 + (kt * 8 + nt) * 512 + lane * 16);
                    ga[nt] = wmma_(a, b, ga[nt]);
                }
            }
            #pragma unroll
            for (int nt = 0; nt < 8; ++nt) {
                #pragma unroll
                for (int r = 0; r < 8; ++r)
                    act0[(r + hi * 8) * ACT_PITCH + nt * 16 + lm] =
                        (_Float16)fmaxf(ga[nt][r] + bG1[nt], 0.f);
            }
        }
        __syncthreads();

        // -------- g2: 16 N-tiles split 2 per wave, coalesced store --------
        {
            v8f z = {};
            v8f o0 = z, o1 = z;
            const int nt0 = wid * 2, nt1 = nt0 + 1;
            #pragma unroll
            for (int kt = 0; kt < 4; ++kt) {
                v16h a = read_a_row(act0 + lm * ACT_PITCH + kt * 32, hi);
                v16h b0 = *(const v16h*)(lh + W_G2 + (kt * 16 + nt0) * 512 + lane * 16);
                v16h b1 = *(const v16h*)(lh + W_G2 + (kt * 16 + nt1) * 512 + lane * 16);
                o0 = wmma_(a, b0, o0);
                o1 = wmma_(a, b1, o1);
            }
            #pragma unroll
            for (int r = 0; r < 8; ++r) {
                int mrow = r + hi * 8;
                int cc   = g * 16 + mrow;
                if (cc < NC) {
                    out[(size_t)cc * 256 + nt0 * 16 + lm] = fmaxf(o0[r] + bG2x0, 0.f);
                    out[(size_t)cc * 256 + nt1 * 16 + lm] = fmaxf(o1[r] + bG2x1, 0.f);
                }
            }
        }
        __syncthreads();
    }
}

extern "C" void kernel_launch(void* const* d_in, const int* in_sizes, int n_in,
                              void* d_out, int out_size, void* d_ws, size_t ws_size,
                              hipStream_t stream) {
    const float* rel  = (const float*)d_in[0];
    const float* feat = (const float*)d_in[1];
    const float* ew1  = (const float*)d_in[2];
    const float* eb1  = (const float*)d_in[3];
    const float* ew2  = (const float*)d_in[4];
    const float* eb2  = (const float*)d_in[5];
    const float* fw1  = (const float*)d_in[6];
    const float* fb1  = (const float*)d_in[7];
    const float* fw2  = (const float*)d_in[8];
    const float* fb2  = (const float*)d_in[9];
    const float* gw1  = (const float*)d_in[10];
    const float* gb1  = (const float*)d_in[11];
    const float* gw2  = (const float*)d_in[12];
    const float* gb2  = (const float*)d_in[13];
    float* out = (float*)d_out;

    const int N       = in_sizes[0] / 3;
    const int NC      = N / PTS;
    const int ngroups = (NC + 15) / 16;

    hipFuncSetAttribute(reinterpret_cast<const void*>(mle_fused),
                        hipFuncAttributeMaxDynamicSharedMemorySize, SMEM_BYTES);
    int grid = ngroups < 1024 ? ngroups : 1024;   // persistent workgroups
    mle_fused<<<grid, BLOCK, SMEM_BYTES, stream>>>(
        rel, feat, ew1, eb1, ew2, eb2, fw1, fb1, fw2, fb2,
        gw1, gb1, gw2, gb2, out, NC, ngroups);
}
